// SparseSpikingLayer_62869731279041
// MI455X (gfx1250) — matrix operand order repaired and checked by reading, not compiled, not run on hardware
//
#include <hip/hip_runtime.h>
#include <hip/hip_bf16.h>
#include <stdint.h>

// CDNA5 WMMA vector types (wave32)
typedef __attribute__((ext_vector_type(16))) _Float16 v16h;
typedef __attribute__((ext_vector_type(8)))  float    v8f;

#define N_IN        8192
#define N_OUT       8192
#define THRESHOLD   1.0f
#define DECAY       0.95122942450071400910f   // exp(-1/20)
#define TRACE_DECAY 0.99f
#define NWAVES      8
#define KSLICE      (N_IN / NWAVES)           // 1024 K per wave
#define KITERS      (KSLICE / 32)             // 32 WMMA steps per wave

// One block = one 16-row output tile. 8 waves split K; W/mask are streamed
// exactly once, fully coalesced. WMMA f16 (f32 accum) does the 16x32 tile
// multiply-reduce; columns of C are identical (B broadcasts x), so lane 0 /
// lane 16 hold the 16 row dot-products at the end.
__global__ __launch_bounds__(256)
void sparse_lif_wmma_kernel(const float* __restrict__ x,
                            const float* __restrict__ W,
                            const float* __restrict__ mem,
                            const float* __restrict__ trace,
                            const float* __restrict__ last,
                            const float* __restrict__ tcur,
                            const uint8_t* __restrict__ mask,
                            float* __restrict__ out)
{
    __shared__ __attribute__((aligned(32))) _Float16 xh[N_IN];  // 16 KB
    __shared__ float red[NWAVES * 16];                          // 512 B

    const int tid = threadIdx.x;

    // Stage x into LDS as f16 once (coalesced, 32 elems/thread).
    for (int i = tid; i < N_IN; i += 256) xh[i] = (_Float16)x[i];
    __syncthreads();

    const int wave = tid >> 5;            // 0..7   : K slice owner
    const int lane = tid & 31;
    const int row  = lane & 15;           // A-matrix M for this lane
    const int sel  = lane >> 4;           // 0: K {0-7,16-23}; 1: K {8-15,24-31}
    const int rowg = blockIdx.x * 16 + row;

    const float*   wrow = W    + (size_t)rowg * N_IN;
    const uint8_t* mrow = mask + (size_t)rowg * N_IN;

    const int kbase0 = wave * KSLICE;
    const int klo    = sel * 8;

    v8f acc = {};

    #pragma unroll 4
    for (int it = 0; it < KITERS; ++it) {
        const int base = kbase0 + it * 32;

        // Weights: lane's two contiguous 8-float chunks of its row.
        const float4* wp = (const float4*)(wrow + base + klo);
        const float4 w0 = wp[0];          // K klo+0..3
        const float4 w1 = wp[1];          // K klo+4..7
        const float4 w2 = wp[4];          // K klo+16..19
        const float4 w3 = wp[5];          // K klo+20..23

        // Mask bytes for the same 16 K positions.
        const uint64_t mlo = *(const uint64_t*)(mrow + base + klo);
        const uint64_t mhi = *(const uint64_t*)(mrow + base + klo + 16);

        // Prefetch 8 iterations (1 KB/row) ahead, guarded against OOB.
        if (it + 8 < KITERS) {
            __builtin_prefetch((const void*)(wrow + base + 8 * 32), 0, 1);
            __builtin_prefetch((const void*)(mrow + base + 8 * 32), 0, 1);
        }

        // Pack masked f16 A-fragment in ISA register order.
        const float wf[16] = {w0.x, w0.y, w0.z, w0.w, w1.x, w1.y, w1.z, w1.w,
                              w2.x, w2.y, w2.z, w2.w, w3.x, w3.y, w3.z, w3.w};
        union { v16h v; _Float16 h[16]; } a;
        #pragma unroll
        for (int i = 0; i < 8; ++i)
            a.h[i] = ((mlo >> (8 * i)) & 1ull) ? (_Float16)wf[i] : (_Float16)0.0f;
        #pragma unroll
        for (int i = 0; i < 8; ++i)
            a.h[8 + i] = ((mhi >> (8 * i)) & 1ull) ? (_Float16)wf[8 + i] : (_Float16)0.0f;

        // B fragment: x broadcast into all 16 columns.
        // Lane holds K = sel*8 + 0..7 and 16 + sel*8 + 0..7 of its column -> two 16B ds loads.
        typedef __attribute__((ext_vector_type(8))) _Float16 v8h;
        const v8h b0 = *(const v8h*)(&xh[base + sel * 8]);
        const v8h b1 = *(const v8h*)(&xh[base + 16 + sel * 8]);
        const v16h b = __builtin_shufflevector(b0, b1, 0, 1, 2, 3, 4, 5, 6, 7, 8, 9, 10, 11, 12, 13, 14, 15);

        acc = __builtin_amdgcn_wmma_f32_16x16x32_f16(
            /*neg_a=*/false, a.v, /*neg_b=*/false, b,
            /*c_mod=*/(short)0, acc, /*reuse_a=*/false, /*reuse_b=*/false);
    }

    // C layout: lane n (0-15) VGPR j = C[M=j][N=n]; lane n+16 = C[M=8+j][N=n].
    // All columns identical -> lane 0 has rows 0-7, lane 16 has rows 8-15.
    if (lane == 0 || lane == 16) {
        #pragma unroll
        for (int j = 0; j < 8; ++j)
            red[wave * 16 + sel * 8 + j] = acc[j];
    }
    __syncthreads();

    // Cross-wave K reduction + LIF pointwise update (one thread per row).
    if (tid < 16) {
        float s = 0.0f;
        #pragma unroll
        for (int w = 0; w < NWAVES; ++w) s += red[w * 16 + tid];

        const int   rg    = blockIdx.x * 16 + tid;
        const float v     = mem[rg] * DECAY + s;
        const float spike = (v >= THRESHOLD) ? 1.0f : 0.0f;
        const bool  fired = spike > 0.0f;

        out[rg]             = spike;                                   // spikes
        out[N_OUT + rg]     = (fired ? 1.0f : trace[rg]) * TRACE_DECAY; // trace
        out[2 * N_OUT + rg] = fired ? tcur[0] : last[rg];               // last t
    }
}

extern "C" void kernel_launch(void* const* d_in, const int* in_sizes, int n_in,
                              void* d_out, int out_size, void* d_ws, size_t ws_size,
                              hipStream_t stream)
{
    (void)in_sizes; (void)n_in; (void)out_size; (void)d_ws; (void)ws_size;
    const float*   x    = (const float*)d_in[0];
    const float*   W    = (const float*)d_in[1];
    const float*   mem  = (const float*)d_in[2];
    const float*   tr   = (const float*)d_in[3];
    const float*   last = (const float*)d_in[4];
    const float*   tcur = (const float*)d_in[5];
    const uint8_t* mask = (const uint8_t*)d_in[6];
    float* out = (float*)d_out;

    sparse_lif_wmma_kernel<<<dim3(N_OUT / 16), dim3(256), 0, stream>>>(
        x, W, mem, tr, last, tcur, mask, out);
}
